// multi_attention_46780783788422
// MI455X (gfx1250) — compile-verified, matrix operand order checked by reading, not run on hardware
//
#include <hip/hip_runtime.h>
#include <hip/hip_bf16.h>

// ---------------------------------------------------------------------------
// Problem constants (reference: B=2, T=4096, D=512, H=8, head_dim=64)
// ---------------------------------------------------------------------------
#define BATCH 2
#define SEQ   4096
#define DM    512
#define NH    8
#define HD    64

typedef __attribute__((ext_vector_type(16))) __bf16 v16bf;
typedef __attribute__((ext_vector_type(8)))  float  v8f;

// fp32 -> bf16 via hardware cvt (v_cvt_pk_bf16_f32 class), not manual RNE.
__device__ __forceinline__ unsigned short f2bf(float x) {
    union { __bf16 h; unsigned short u; } c;
    c.h = (__bf16)x;
    return c.u;
}
__device__ __forceinline__ unsigned pack_bf16(float x, float y) {
    return (unsigned)f2bf(x) | ((unsigned)f2bf(y) << 16);
}

// ---------------------------------------------------------------------------
// Async global->LDS copies (CDNA5 GLOBAL_LOAD_ASYNC_TO_LDS_B128, ASYNCcnt).
// Builtin signature (from hipcc diagnostic): param1 = int4* in the global
// address space, param2 = LDS-side pointer, then imm offset + imm cpol.
// LDS offset = low 32 bits of the flat address (ISA 10.2); global AS1 address
// equals the flat VA. Integer casts avoid addrspacecast legality questions.
// Falls back to synchronous uint4 staging if the builtins are absent.
// ---------------------------------------------------------------------------
#if __has_builtin(__builtin_amdgcn_global_load_async_to_lds_b128) && \
    __has_builtin(__builtin_amdgcn_s_wait_asynccnt)
#define HAVE_ASYNC_LDS 1
typedef int v4i __attribute__((vector_size(16)));
typedef __attribute__((address_space(1))) v4i* as1_v4i;
typedef __attribute__((address_space(3))) v4i* as3_v4i;
__device__ __forceinline__ void cp_b128(void* lds_dst, const void* gsrc) {
    __builtin_amdgcn_global_load_async_to_lds_b128(
        (as1_v4i)(unsigned long long)(size_t)gsrc,
        (as3_v4i)(unsigned)(size_t)lds_dst, 0, 0);
}
__device__ __forceinline__ void async_wait0() {
    __builtin_amdgcn_s_wait_asynccnt(0);
}
#else
#define HAVE_ASYNC_LDS 0
__device__ __forceinline__ void cp_b128(void* lds_dst, const void* gsrc) {
    *(uint4*)lds_dst = *(const uint4*)gsrc;
}
__device__ __forceinline__ void async_wait0() {}
#endif

// Load a 16x32 bf16 fragment (A-style: per-lane row fixed, packed K pairs).
// ISA 16-bit layout: lane L -> row = caller-provided; half = L/16;
//   VGPR i<4 : K = 2i + 8*half (pair), VGPR i>=4 : K = 16 + 2(i-4) + 8*half.
// B-fragments mirror this with lane = N, so the same loader works on an
// [n][k] row-major tile.
__device__ __forceinline__ v16bf load_frag(const unsigned short* base, int ld,
                                           int row, int koff) {
    union { v16bf v; unsigned u[8]; } f;
    const int half = (threadIdx.x >> 4) & 1;
    const unsigned short* r = base + row * ld + koff + 8 * half;
#pragma unroll
    for (int i = 0; i < 4; ++i)
        f.u[i] = *(const unsigned int*)(r + 2 * i);
#pragma unroll
    for (int i = 0; i < 4; ++i)
        f.u[4 + i] = *(const unsigned int*)(r + 16 + 2 * i);
    return f.v;
}

// Constant all-ones bf16 fragment (for tensor-core row sums).
__device__ __forceinline__ v16bf ones_frag() {
    union { v16bf v; unsigned u[8]; } f;
#pragma unroll
    for (int i = 0; i < 8; ++i) f.u[i] = 0x3F803F80u;  // {1.0bf16, 1.0bf16}
    return f.v;
}

__device__ __forceinline__ v8f wmma_bf16(v16bf a, v16bf b, v8f c) {
    return __builtin_amdgcn_wmma_f32_16x16x32_bf16(
        /*neg_a=*/false, a, /*neg_b=*/false, b,
        /*c_mod=*/(short)0, c, /*reuse_a=*/false, /*reuse_b=*/false);
}

// Cross-lane xor within 16-lane rows. Prefer v_permlane16_b32 (pure VALU, no
// DS round-trip); fall back to __shfl_xor (ds_bpermute) if unavailable.
#if __has_builtin(__builtin_amdgcn_permlane16)
__device__ __forceinline__ float xor16(float x, int step) {
    unsigned v;
    __builtin_memcpy(&v, &x, 4);
    unsigned r;
    switch (step) {
    case 1:  r = __builtin_amdgcn_permlane16(v, v, 0x67452301u, 0xEFCDAB89u, false, false); break;
    case 2:  r = __builtin_amdgcn_permlane16(v, v, 0x54761032u, 0xDCFE98BAu, false, false); break;
    case 4:  r = __builtin_amdgcn_permlane16(v, v, 0x32107654u, 0xBA98FEDCu, false, false); break;
    default: r = __builtin_amdgcn_permlane16(v, v, 0xFEDCBA98u, 0x76543210u, false, false); break;
    }
    float out;
    __builtin_memcpy(&out, &r, 4);
    return out;
}
#else
__device__ __forceinline__ float xor16(float x, int step) {
    return __shfl_xor(x, step, 32);
}
#endif

__device__ __forceinline__ float exp2_fast(float x) {
    return __builtin_amdgcn_exp2f(x);   // v_exp_f32 (base-2 on AMD)
}

// ---------------------------------------------------------------------------
// Generic 64x64-tile GEMM, BK=64:  C[M,N] = (A[M,K] * W[K,N] + bias) * oscale
// 128 threads (4 waves); wave w computes a 16x64 strip (4 WMMA accumulators).
// Staging must convert fp32->bf16 in flight, so it uses VGPR loads (async
// byte copies cannot convert).
// MODE 0: write bf16, head-split  [B,H,T,HD]      (Q, K projections)
// MODE 1: write bf16, transposed  [B,H,HD,T]      (V projection)
// MODE 2: write fp32 row-major    [M,N]           (output projection)
// ---------------------------------------------------------------------------
template <int MODE>
__global__ void gemm64(const float* __restrict__ A, const float* __restrict__ W,
                       const float* __restrict__ bias, void* __restrict__ outp,
                       int M, int N, int K, float oscale) {
    __shared__ unsigned short As[64][72];   // [m][k]   (row stride 144B = 16*9)
    __shared__ unsigned short Ws[64][72];   // [n][k]   transposed W tile

    const int tid  = threadIdx.x;
    const int wave = tid >> 5;
    const int lane = tid & 31;
    const int lh   = lane >> 4;
    const int ln   = lane & 15;

    const int n0 = blockIdx.x * 64;
    const int m0 = blockIdx.y * 64;

    v8f acc[4] = {v8f{0.f}, v8f{0.f}, v8f{0.f}, v8f{0.f}};

    for (int k0 = 0; k0 < K; k0 += 64) {
        __syncthreads();
        // Stage A tile 64x64 (fp32 -> bf16 packed pairs), float4-coalesced.
        for (int i = tid; i < 1024; i += 128) {
            const int r = i >> 4, cg = (i & 15) * 4;
            float4 f = *(const float4*)(A + (size_t)(m0 + r) * K + k0 + cg);
            unsigned* d = (unsigned*)&As[r][cg];
            d[0] = pack_bf16(f.x, f.y);
            d[1] = pack_bf16(f.z, f.w);
        }
        // Stage W tile 64x64 transposed into [n][k].
        for (int i = tid; i < 1024; i += 128) {
            const int kr = i >> 4, cg = (i & 15) * 4;
            float4 f = *(const float4*)(W + (size_t)(k0 + kr) * N + n0 + cg);
            Ws[cg + 0][kr] = f2bf(f.x);
            Ws[cg + 1][kr] = f2bf(f.y);
            Ws[cg + 2][kr] = f2bf(f.z);
            Ws[cg + 3][kr] = f2bf(f.w);
        }
        __syncthreads();

        v16bf a0 = load_frag(&As[0][0], 72, wave * 16 + ln, 0);
        v16bf a1 = load_frag(&As[0][0], 72, wave * 16 + ln, 32);
#pragma unroll
        for (int nt = 0; nt < 4; ++nt) {
            v16bf b0 = load_frag(&Ws[0][0], 72, nt * 16 + ln, 0);
            acc[nt] = wmma_bf16(a0, b0, acc[nt]);
            v16bf b1 = load_frag(&Ws[0][0], 72, nt * 16 + ln, 32);
            acc[nt] = wmma_bf16(a1, b1, acc[nt]);
        }
    }

    // Epilogue. C/D layout: VGPR j, lane L -> row M = j + 8*(L/16), col N = L%16.
#pragma unroll
    for (int nt = 0; nt < 4; ++nt) {
        const int n = n0 + nt * 16 + ln;
        const float bv = bias[n];
#pragma unroll
        for (int j = 0; j < 8; ++j) {
            const int m = m0 + wave * 16 + 8 * lh + j;
            const float val = (acc[nt][j] + bv) * oscale;
            if (MODE == 2) {
                ((float*)outp)[(size_t)m * N + n] = val;
            } else {
                const int b = m >> 12;        // T = 4096
                const int t = m & (SEQ - 1);
                const int h = n >> 6;         // HD = 64
                const int dd = n & (HD - 1);
                const unsigned short o = f2bf(val);
                if (MODE == 0)
                    ((unsigned short*)outp)[(((size_t)(b * NH + h) * SEQ) + t) * HD + dd] = o;
                else
                    ((unsigned short*)outp)[(((size_t)(b * NH + h) * HD) + dd) * SEQ + t] = o;
            }
        }
    }
}

// ---------------------------------------------------------------------------
// Flash attention: one block = (b, h, 64-query tile); 4 waves, 16 rows each.
// qh,kh : bf16 [B,H,T,HD]   vt : bf16 [B,H,HD,T]   attn : fp32 [B,T,D]
// K/V tiles are DOUBLE-BUFFERED in LDS and filled with async global->LDS
// copies: tile kt+1 streams in (ASYNCcnt) while tile kt runs 18 WMMAs +
// softmax; one s_wait_asynccnt + one barrier per k-tile.
// Q was pre-scaled by (1/sqrt(HD))*log2(e): scores are in the log2 domain,
// softmax uses raw v_exp_f32. Row sums of P come from the tensor core
// (P x ones); row max uses a permlane16 xor-butterfly (pure VALU).
// ---------------------------------------------------------------------------
__global__ void flash_attn(const unsigned short* __restrict__ qh,
                           const unsigned short* __restrict__ kh,
                           const unsigned short* __restrict__ vt,
                           float* __restrict__ attn) {
    __shared__ unsigned short Qs[64][72];         // [q][d]  (row 144B = 16*9)
    __shared__ unsigned short Ks[2][64][72];      // [key][d], double-buffered
    __shared__ unsigned short Vt[2][64][72];      // [d][key], double-buffered
    __shared__ unsigned short Ps[4][16][72];      // per-wave P [q][key]

    const int qt = blockIdx.x;      // query tile (64 rows)
    const int h  = blockIdx.y;
    const int b  = blockIdx.z;
    const int tid  = threadIdx.x;
    const int wave = tid >> 5;
    const int lane = tid & 31;
    const int lh   = lane >> 4;
    const int ln   = lane & 15;

    const size_t headQK = ((size_t)(b * NH + h) * SEQ) * HD;
    const size_t headV  = ((size_t)(b * NH + h) * HD) * SEQ;

    auto stage_kv = [&](int buf, int k0) {
        for (int i = tid; i < 512; i += 128) {
            const int r = i >> 3, c = (i & 7) * 8;
            cp_b128(&Ks[buf][r][c], kh + headQK + (size_t)(k0 + r) * HD + c);
            cp_b128(&Vt[buf][r][c], vt + headV + (size_t)r * SEQ + k0 + c);
        }
    };

    // Stage Q tile and the first K/V tile together (overlapped if async).
    for (int i = tid; i < 512; i += 128) {
        const int r = i >> 3, c = (i & 7) * 8;
        cp_b128(&Qs[r][c], qh + headQK + (size_t)(qt * 64 + r) * HD + c);
    }
    stage_kv(0, 0);
    async_wait0();
    __syncthreads();

    const v16bf aq0 = load_frag(&Qs[0][0], 72, wave * 16 + ln, 0);
    const v16bf aq1 = load_frag(&Qs[0][0], 72, wave * 16 + ln, 32);
    const v16bf vones = ones_frag();

    v8f   acc[4] = {v8f{0.f}, v8f{0.f}, v8f{0.f}, v8f{0.f}};
    float m_run[8], l_run[8];
#pragma unroll
    for (int j = 0; j < 8; ++j) { m_run[j] = -1e30f; l_run[j] = 0.f; }

    for (int kt = 0; kt <= qt; ++kt) {
        const int k0  = kt * 64;
        const int cur = kt & 1;

        // Stream the NEXT tile into the other buffer while computing this one.
        if (kt < qt) stage_kv(cur ^ 1, k0 + 64);
        if (kt + 2 <= qt)  // and hint the tile after that toward the caches
            __builtin_prefetch(kh + headQK + (size_t)(k0 + 128) * HD + lane * 8, 0, 1);

        // Scores S (log2 domain), 4 tiles of 16 keys.
        v8f S[4];
#pragma unroll
        for (int nt = 0; nt < 4; ++nt) {
            v8f s = v8f{0.f};
            v16bf bk0 = load_frag(&Ks[cur][0][0], 72, nt * 16 + ln, 0);
            s = wmma_bf16(aq0, bk0, s);
            v16bf bk1 = load_frag(&Ks[cur][0][0], 72, nt * 16 + ln, 32);
            s = wmma_bf16(aq1, bk1, s);
            if (kt == qt) {  // causal mask on the diagonal tile
                const int kglob = k0 + nt * 16 + ln;
#pragma unroll
                for (int j = 0; j < 8; ++j) {
                    const int qglob = qt * 64 + wave * 16 + 8 * lh + j;
                    if (kglob > qglob) s[j] = -1e30f;
                }
            }
            S[nt] = s;
        }

        // Row max: tile-local max + 4 permlane16 xor-butterfly steps.
        float mx[8];
#pragma unroll
        for (int j = 0; j < 8; ++j) {
            float m = S[0][j];
            m = fmaxf(m, S[1][j]); m = fmaxf(m, S[2][j]); m = fmaxf(m, S[3][j]);
            mx[j] = m;
        }
#pragma unroll
        for (int off = 1; off < 16; off <<= 1) {
            float t[8];
#pragma unroll
            for (int j = 0; j < 8; ++j) t[j] = xor16(mx[j], off);
#pragma unroll
            for (int j = 0; j < 8; ++j) mx[j] = fmaxf(mx[j], t[j]);
        }

        float alpha[8];
#pragma unroll
        for (int j = 0; j < 8; ++j) {
            const float mnew = fmaxf(m_run[j], mx[j]);
            alpha[j] = exp2_fast(m_run[j] - mnew);
            m_run[j] = mnew;
        }
        // Rescale O accumulator BEFORE adding this tile's PV contribution.
#pragma unroll
        for (int nt = 0; nt < 4; ++nt)
#pragma unroll
            for (int j = 0; j < 8; ++j) acc[nt][j] *= alpha[j];

        // P = exp2(S - m); stage bf16 P for the A-fragment layout
        // (wave-private LDS region: no barrier needed, DS is in-order per wave).
#pragma unroll
        for (int nt = 0; nt < 4; ++nt) {
#pragma unroll
            for (int j = 0; j < 8; ++j) {
                const float p = exp2_fast(S[nt][j] - m_run[j]);
                Ps[wave][8 * lh + j][nt * 16 + ln] = f2bf(p);
            }
        }

        // O += P*V, and row sums l += P*1 on the tensor core.
        v8f lacc = v8f{0.f};
#pragma unroll
        for (int ks = 0; ks < 2; ++ks) {
            v16bf pf = load_frag(&Ps[wave][0][0], 72, ln, 32 * ks);
            lacc = wmma_bf16(pf, vones, lacc);
#pragma unroll
            for (int nt = 0; nt < 4; ++nt) {
                v16bf bv = load_frag(&Vt[cur][0][0], 72, nt * 16 + ln, 32 * ks);
                acc[nt] = wmma_bf16(pf, bv, acc[nt]);
            }
        }
#pragma unroll
        for (int j = 0; j < 8; ++j) l_run[j] = l_run[j] * alpha[j] + lacc[j];

        // Next tile must be fully in LDS (and everyone done) before use.
        if (kt < qt) {
            async_wait0();
            __syncthreads();
        }
    }

    // Normalize (v_rcp once per row) and write fp32 [B,T,D] at col block h*64.
    float rinv[8];
#pragma unroll
    for (int j = 0; j < 8; ++j) rinv[j] = __builtin_amdgcn_rcpf(l_run[j]);
#pragma unroll
    for (int nt = 0; nt < 4; ++nt) {
#pragma unroll
        for (int j = 0; j < 8; ++j) {
            const int t   = qt * 64 + wave * 16 + 8 * lh + j;
            const int col = h * HD + nt * 16 + ln;
            attn[((size_t)b * SEQ + t) * DM + col] = acc[nt][j] * rinv[j];
        }
    }
}

// ---------------------------------------------------------------------------
extern "C" void kernel_launch(void* const* d_in, const int* in_sizes, int n_in,
                              void* d_out, int out_size, void* d_ws, size_t ws_size,
                              hipStream_t stream) {
    const float* v  = (const float*)d_in[0];
    const float* k  = (const float*)d_in[1];
    const float* q  = (const float*)d_in[2];
    const float* wq = (const float*)d_in[3];
    const float* bq = (const float*)d_in[4];
    const float* wk = (const float*)d_in[5];
    const float* bk = (const float*)d_in[6];
    const float* wv = (const float*)d_in[7];
    const float* bv = (const float*)d_in[8];
    const float* wo = (const float*)d_in[9];
    const float* bo = (const float*)d_in[10];

    const int M = BATCH * SEQ;          // 8192
    const int N = DM;                   // 512
    const int K = DM;                   // 512
    const size_t headElems = (size_t)BATCH * NH * SEQ * HD;  // 4,194,304

    char* ws = (char*)d_ws;
    unsigned short* qh   = (unsigned short*)(ws);
    unsigned short* kh   = (unsigned short*)(ws + headElems * 2);
    unsigned short* vt   = (unsigned short*)(ws + headElems * 4);
    float*          attn = (float*)         (ws + headElems * 6);

    // Q is pre-scaled so attention scores are in the log2 domain:
    // (1/sqrt(64)) * log2(e)
    const float qscale = 0.125f * 1.44269504088896340736f;

    dim3 gGemm(N / 64, M / 64);         // 8 x 128 blocks
    dim3 bGemm(128);
    gemm64<0><<<gGemm, bGemm, 0, stream>>>(q, wq, bq, (void*)qh, M, N, K, qscale);
    gemm64<0><<<gGemm, bGemm, 0, stream>>>(k, wk, bk, (void*)kh, M, N, K, 1.0f);
    gemm64<1><<<gGemm, bGemm, 0, stream>>>(v, wv, bv, (void*)vt, M, N, K, 1.0f);

    dim3 gAttn(SEQ / 64, NH, BATCH);    // 64 x 8 x 2 blocks
    flash_attn<<<gAttn, dim3(128), 0, stream>>>(qh, kh, vt, attn);

    gemm64<2><<<gGemm, bGemm, 0, stream>>>(attn, wo, bo, d_out, M, N, K, 1.0f);
}
